// GIN_Net_65163243815282
// MI455X (gfx1250) — matrix operand order repaired and checked by reading, not compiled, not run on hardware
//
#include <hip/hip_runtime.h>
#include <hip/hip_bf16.h>

// ---------------- types ----------------
typedef __attribute__((ext_vector_type(16))) __bf16 v16bf;
typedef __attribute__((ext_vector_type(8)))  __bf16 v8bf;
typedef __attribute__((ext_vector_type(8)))  float  v8f;

#define GN 100000     // nodes
#define GE 1600000    // edges
#define GF 128        // features / hidden
#define GC 64         // classes
#define PITCH 136     // LDS row pitch in bf16 (272B = 68 banks -> conflict-free, 16B aligned)

// ---------------- helpers ----------------
__device__ __forceinline__ __bf16 f2bf(float f) {
    union { float f; unsigned u; } a; a.f = f;
    unsigned r = a.u + 0x7FFFu + ((a.u >> 16) & 1u);
    union { unsigned short s; __bf16 b; } o; o.s = (unsigned short)(r >> 16);
    return o.b;
}

// A-fragment: 16-bit A 16x32 layout -> per lane: elems 0..7 = K[half*8 .. +7],
// elems 8..15 = K[half*8+16 .. +23]  => two 16B loads with a 16-element gap.
__device__ __forceinline__ v16bf ld_frag_a(const __bf16* p) {
    v8bf lo = *(const v8bf*)(p);
    v8bf hi = *(const v8bf*)(p + 16);
    return __builtin_shufflevector(lo, hi, 0,1,2,3,4,5,6,7,8,9,10,11,12,13,14,15);
}

// B-fragment: per lane col = lane%16, elems 0..7 = K[half*8 .. +7], elems 8..15 =
// K[half*8+16 .. +23] from transposed bf16 weights (wT[n][k]) => two 16B global loads.
__device__ __forceinline__ v16bf ld_frag_b(const __bf16* p) {
    v8bf lo = *(const v8bf*)(p);
    v8bf hi = *(const v8bf*)(p + 16);
    return __builtin_shufflevector(lo, hi, 0,1,2,3,4,5,6,7,8,9,10,11,12,13,14,15);
}

// ---------------- small utility kernels ----------------
// f32 [K][Nout] -> bf16 transposed [Nout][K] (K always 128)
__global__ void k_w_to_bf16T(const float* __restrict__ w, __bf16* __restrict__ wT,
                             int K, int Nout) {
    int idx = blockIdx.x * blockDim.x + threadIdx.x;
    if (idx >= K * Nout) return;
    int k = idx / Nout, n = idx % Nout;
    wT[n * K + k] = f2bf(w[idx]);
}

__global__ void k_copy_f4(const float4* __restrict__ src, float4* __restrict__ dst, int n4) {
    int i = blockIdx.x * blockDim.x + threadIdx.x;
    if (i < n4) dst[i] = src[i];
}

// ---------------- scatter-add (segment_sum) ----------------
// one lane per (edge, 4-feature chunk): float4 gather + 4x f32 atomic add.
// agg buffer (51.2 MB) is L2-resident on MI455X (192 MB L2).
__global__ void k_scatter_add(const float* __restrict__ feat,
                              const int* __restrict__ ei,
                              float* __restrict__ agg) {
    int tid = blockIdx.x * blockDim.x + threadIdx.x;
    int e = tid >> 5;
    if (e >= GE) return;
    int chunk = tid & 31;
    int src = ei[e];
    int dst = ei[GE + e];
    const float4 v = *(const float4*)(feat + (size_t)src * GF + chunk * 4);
    float* p = agg + (size_t)dst * GF + chunk * 4;
    atomicAdd(p + 0, v.x);
    atomicAdd(p + 1, v.y);
    atomicAdd(p + 2, v.z);
    atomicAdd(p + 3, v.w);
}

// ---------------- fused 2-GEMM MLP (GIN conv MLP) ----------------
// block = 256 threads (8 waves), each block handles 128 rows.
// GEMM1: [128,128]x[128,128] bf16 WMMA -> ReLU -> (LDS) -> GEMM2: [128,128]x[128,NOUT2]
// Final activation: ELU (layer 1) or identity (layer 2).
template <int NOUT2, bool ELU_FINAL>
__global__ void __launch_bounds__(256)
k_gin_mlp(const float* __restrict__ agg,
          const __bf16* __restrict__ wT1, const float* __restrict__ b1,
          const __bf16* __restrict__ wT2, const float* __restrict__ b2,
          float* __restrict__ out) {
    __shared__ __bf16 sA[128 * PITCH];   // 34816 B: input tile, reused for mid activations

    const int tid  = threadIdx.x;
    const int wave = tid >> 5;
    const int lane = tid & 31;
    const int half = lane >> 4;          // K-half select for WMMA fragments
    const int lcol = lane & 15;          // col (B/C/D) / row (A) within tile
    const int rowBase = blockIdx.x * 128;

    // ---- stage agg tile into LDS as bf16 (zero-pad rows >= N) ----
    {
        const int r  = tid >> 1;               // 0..127
        const int c0 = (tid & 1) * 64;
        const int grow = rowBase + r;
        #pragma unroll
        for (int i = 0; i < 16; ++i) {
            const int c = c0 + i * 4;
            float4 v = make_float4(0.f, 0.f, 0.f, 0.f);
            if (grow < GN) v = *(const float4*)(agg + (size_t)grow * GF + c);
            __bf16* d = &sA[r * PITCH + c];
            d[0] = f2bf(v.x); d[1] = f2bf(v.y); d[2] = f2bf(v.z); d[3] = f2bf(v.w);
        }
    }
    __syncthreads();

    const int wr = wave * 16;                  // wave's 16-row strip in tile

    // ---- GEMM1: 8 column tiles of 16, K = 128 in 4 steps of 32 ----
    v8f acc[8];
    #pragma unroll
    for (int nt = 0; nt < 8; ++nt) {
        const float bias = b1[nt * 16 + lcol];
        v8f c;
        #pragma unroll
        for (int v = 0; v < 8; ++v) c[v] = bias;
        #pragma unroll
        for (int kt = 0; kt < 4; ++kt) {
            v16bf a = ld_frag_a(&sA[(wr + lcol) * PITCH + kt * 32 + half * 8]);
            v16bf b = ld_frag_b(&wT1[(size_t)(nt * 16 + lcol) * GF + kt * 32 + half * 8]);
            c = __builtin_amdgcn_wmma_f32_16x16x32_bf16(
                    false, a, false, b, (short)0, c, false, false);
        }
        acc[nt] = c;
    }
    __syncthreads();

    // ---- ReLU -> bf16 mid activations back into LDS ----
    #pragma unroll
    for (int nt = 0; nt < 8; ++nt) {
        const int col = nt * 16 + lcol;
        #pragma unroll
        for (int v = 0; v < 8; ++v) {
            const int m = wr + half * 8 + v;   // C/D layout: M = vgpr + (lane/16)*8
            float x = acc[nt][v];
            x = x > 0.f ? x : 0.f;
            sA[m * PITCH + col] = f2bf(x);
        }
    }
    __syncthreads();

    // ---- GEMM2: NOUT2/16 column tiles ----
    #pragma unroll
    for (int nt = 0; nt < NOUT2 / 16; ++nt) {
        const float bias = b2[nt * 16 + lcol];
        v8f c;
        #pragma unroll
        for (int v = 0; v < 8; ++v) c[v] = bias;
        #pragma unroll
        for (int kt = 0; kt < 4; ++kt) {
            v16bf a = ld_frag_a(&sA[(wr + lcol) * PITCH + kt * 32 + half * 8]);
            v16bf b = ld_frag_b(&wT2[(size_t)(nt * 16 + lcol) * GF + kt * 32 + half * 8]);
            c = __builtin_amdgcn_wmma_f32_16x16x32_bf16(
                    false, a, false, b, (short)0, c, false, false);
        }
        // ---- final activation + f32 store ----
        #pragma unroll
        for (int v = 0; v < 8; ++v) {
            const int m = rowBase + wr + half * 8 + v;
            float x = c[v];
            if (ELU_FINAL) x = x > 0.f ? x : (__expf(x) - 1.0f);
            if (m < GN) out[(size_t)m * NOUT2 + nt * 16 + lcol] = x;
        }
    }
}

// ---------------- launcher ----------------
extern "C" void kernel_launch(void* const* d_in, const int* in_sizes, int n_in,
                              void* d_out, int out_size, void* d_ws, size_t ws_size,
                              hipStream_t stream) {
    const float* x    = (const float*)d_in[0];
    const float* Q    = (const float*)d_in[1];
    const float* w1a  = (const float*)d_in[2];
    const float* b1a  = (const float*)d_in[3];
    const float* w1b  = (const float*)d_in[4];
    const float* b1b  = (const float*)d_in[5];
    const float* w2a  = (const float*)d_in[6];
    const float* b2a  = (const float*)d_in[7];
    const float* w2b  = (const float*)d_in[8];
    const float* b2b  = (const float*)d_in[9];
    const int*   ei   = (const int*)d_in[10];
    float* out = (float*)d_out;

    // workspace layout
    float*  agg  = (float*)d_ws;                 // N*128 f32 (reused for both layers)
    float*  h    = agg + (size_t)GN * GF;        // N*128 f32
    __bf16* w1aT = (__bf16*)(h + (size_t)GN * GF);
    __bf16* w1bT = w1aT + GF * GF;
    __bf16* w2aT = w1bT + GF * GF;
    __bf16* w2bT = w2aT + GF * GF;               // 128*64

    // 1) weights -> transposed bf16 (tiny, L2-resident afterwards)
    k_w_to_bf16T<<<(GF * GF + 255) / 256, 256, 0, stream>>>(w1a, w1aT, GF, GF);
    k_w_to_bf16T<<<(GF * GF + 255) / 256, 256, 0, stream>>>(w1b, w1bT, GF, GF);
    k_w_to_bf16T<<<(GF * GF + 255) / 256, 256, 0, stream>>>(w2a, w2aT, GF, GF);
    k_w_to_bf16T<<<(GF * GC + 255) / 256, 256, 0, stream>>>(w2b, w2bT, GF, GC);

    const int n4feat = GN * GF / 4;              // 3.2M float4
    const int scatterThreads = GE * 32;          // one lane per (edge, 4-feat chunk)

    // 2) agg = x ; agg += scatter(x)
    k_copy_f4<<<(n4feat + 255) / 256, 256, 0, stream>>>((const float4*)x, (float4*)agg, n4feat);
    k_scatter_add<<<(scatterThreads + 255) / 256, 256, 0, stream>>>(x, ei, agg);

    // 3) h = ELU(MLP1(agg))
    k_gin_mlp<GF, true><<<(GN + 127) / 128, 256, 0, stream>>>(agg, w1aT, b1a, w1bT, b1b, h);

    // 4) agg = h ; agg += scatter(h)
    k_copy_f4<<<(n4feat + 255) / 256, 256, 0, stream>>>((const float4*)h, (float4*)agg, n4feat);
    k_scatter_add<<<(scatterThreads + 255) / 256, 256, 0, stream>>>(h, ei, agg);

    // 5) out = MLP2(agg)  -> first N*C floats of d_out
    k_gin_mlp<GC, false><<<(GN + 127) / 128, 256, 0, stream>>>(agg, w2aT, b2a, w2bT, b2b, out);

    // 6) Q pass-through -> remaining floats of d_out
    const int n4q = (GE / 2) * GC / 4;           // 12.8M float4
    k_copy_f4<<<(n4q + 255) / 256, 256, 0, stream>>>((const float4*)Q,
                                                     (float4*)(out + (size_t)GN * GC), n4q);
}
